// StgcnLayer_81123342287292
// MI455X (gfx1250) — matrix operand order repaired and checked by reading, not compiled, not run on hardware
//
#include <hip/hip_runtime.h>

typedef unsigned short u16_t;
typedef unsigned int   u32_t;

typedef __attribute__((ext_vector_type(16))) __bf16 v16bf;
typedef __attribute__((ext_vector_type(8)))  float  v8f;
typedef __attribute__((ext_vector_type(8)))  u32_t  v8u;
typedef __attribute__((ext_vector_type(4)))  u32_t  v4u;

#define CIN   128
#define COUT  128
#define NP    3
#define NV    25
#define NL    300
#define NB    32
#define SP    7500             // L*V spatial columns per (n, channel)
#define LT    16               // l-values per tile
#define ST    (LT*NV)          // 400 spatial columns per tile

// LDS layout (u16 element offsets); phase-2 a' region aliases the x panel
#define XSP   136              // x panel pitch: 256B row + 16B TDM pad = 272B (16B-aligned)
#define WSP   136              // w panel pitch (same TDM pad config)
#define AHP   104              // Ahat / a' pitch (208B)
#define WSOFF (ST*XSP)         // 54400
#define AHOFF (WSOFF + 48*WSP) // 60928
#define SMEMU (AHOFF + 32*AHP) // 64256 u16 = 128512 B

// ---------- helpers ----------

static __device__ __forceinline__ u16_t f2bf(float f) {
  u32_t u = __builtin_bit_cast(u32_t, f);
  u = u + 0x7FFFu + ((u >> 16) & 1u);
  return (u16_t)(u >> 16);
}

static __device__ __forceinline__ unsigned to_lds(const void* p) {
  // generic->LDS addrspacecast is a 32-bit truncate on amdgcn
  return (unsigned)(unsigned long long)p;
}

// CDNA5 async DMA: one instruction moves 16B per active lane into LDS (ASYNCcnt)
static __device__ __forceinline__ void async_copy_b128(unsigned lds_dst, const void* src) {
  asm volatile("global_load_async_to_lds_b128 %0, %1, off"
               :: "v"(lds_dst), "v"(src) : "memory");
}
static __device__ __forceinline__ void wait_async0() {
  asm volatile("s_wait_asynccnt 0x0" ::: "memory");
}

// CDNA5 Tensor Data Mover: 2D tile (rows x 128 bf16) -> LDS, with LDS row padding
// 256B data + 16B pad per row (pad_interval=5 -> 64 DWORDs, pad_amount=3 -> 4 DWORDs).
// Issue from ONE wave only (TDM ignores EXEC); completion via TENSORcnt.
static __device__ __forceinline__ void tdm_load_2d(unsigned lds_addr, const void* gptr,
                                                   unsigned rows, unsigned row_stride_elems) {
  unsigned long long ga = (unsigned long long)gptr;
  v4u g0;
  g0[0] = 1u;                                   // count=1, is_restore=0, gather off
  g0[1] = lds_addr;                             // LDS byte address
  g0[2] = (u32_t)ga;                            // global_addr[31:0]
  g0[3] = (u32_t)(ga >> 32) | (2u << 30);       // global_addr[56:32] | type=2 (image)
  v8u g1;
  g1[0] = (1u << 16) | (1u << 20) | (5u << 22) | (3u << 25); // data_size=2B, pad_enable
  g1[1] = (128u & 0xFFFFu) << 16;               // tensor_dim0 = 128 (lo16)
  g1[2] = (rows & 0xFFFFu) << 16;               // dim0 hi=0 | tensor_dim1 lo16 = rows
  g1[3] = (128u << 16);                         // dim1 hi=0 | tile_dim0 = 128
  g1[4] = rows & 0xFFFFu;                       // tile_dim1 = rows, tile_dim2 = 0
  g1[5] = row_stride_elems;                     // tensor_dim0_stride lo32 (elems)
  g1[6] = 0u;                                   // stride hi | dim1_stride lo (unused, 2D)
  g1[7] = 0u;
  asm volatile("tensor_load_to_lds %0, %1" :: "s"(g0), "s"(g1) : "memory");
}

static __device__ __forceinline__ v16bf ld_frag2(const u16_t* p0, const u16_t* p1) {
  uint4 lo = *(const uint4*)p0;
  uint4 hi = *(const uint4*)p1;
  v8u u;
  u[0]=lo.x; u[1]=lo.y; u[2]=lo.z; u[3]=lo.w;
  u[4]=hi.x; u[5]=hi.y; u[6]=hi.z; u[7]=hi.w;
  return __builtin_bit_cast(v16bf, u);
}

// ---------- kernel 0a: W fp32 -> bf16 ----------

__global__ __launch_bounds__(256)
void convw_kernel(const float* __restrict__ W, u16_t* __restrict__ Wb, int n) {
  int i = blockIdx.x * 256 + threadIdx.x;
  if (i < n) Wb[i] = f2bf(W[i]);
}

// ---------- kernel 0b: Ahat (graph operand + bias row-sums), bf16, col-major K ----------

__global__ __launch_bounds__(256)
void ahat_kernel(const float* __restrict__ A, u16_t* __restrict__ AhatG) {
  int i = blockIdx.x * 256 + threadIdx.x;
  if (i >= 32 * 96) return;
  int col = i / 96, K = i % 96;
  float val = 0.f;
  if (col < NV) {
    if (K < 75) {
      int p = K / 25, w = K % 25;
      val = A[(p * NV + w) * NV + col];
    } else if (K < 78) {
      int p = K - 75;
      float s = 0.f;
      for (int w = 0; w < NV; ++w) s += A[(p * NV + w) * NV + col];
      val = s;
    }
  }
  AhatG[col * 96 + K] = f2bf(val);
}

// ---------- kernel 0c: x fp32 [n][k][s] -> bf16 xT [n][s][k] ----------

__global__ __launch_bounds__(256)
void xpose_kernel(const float* __restrict__ x, u16_t* __restrict__ xT) {
  __shared__ u16_t tile[64][136];
  const int n = blockIdx.y, s0 = blockIdx.x * 64;
  for (int i = threadIdx.x; i < 64 * CIN; i += 256) {
    int k = i / 64, s = i % 64;
    int sg = s0 + s;
    float v = (sg < SP) ? x[((size_t)n * CIN + k) * SP + sg] : 0.f;
    tile[s][k] = f2bf(v);
  }
  __syncthreads();
  for (int i = threadIdx.x; i < 64 * 16; i += 256) {
    int s = i / 16, part = i % 16;
    int sg = s0 + s;
    if (sg < SP)
      *(uint4*)&xT[((size_t)n * SP + sg) * CIN + part * 8] = *(const uint4*)&tile[s][part * 8];
  }
}

// ---------- kernel 1: TDM-staged bf16 WMMA GEMM + WMMA graph mix -> e ----------

__global__ __launch_bounds__(256)
void stgcn_gemm_kernel(const u16_t* __restrict__ xT,
                       const u16_t* __restrict__ Wb,
                       const u16_t* __restrict__ AhatG,
                       const float* __restrict__ bias,
                       float* __restrict__ e) {
  __shared__ __align__(16) u16_t smem[SMEMU];
  u16_t* xs   = smem;                            // [s][k] : 400 x 128 bf16, pitch XSP
  u16_t* wst  = smem + WSOFF;                    // [p*16+m][k] : 48 x 128 bf16, pitch WSP
  u16_t* ahat = smem + AHOFF;                    // [col][K] : 32 x 96 bf16, pitch AHP
  u16_t* as2  = smem;                            // phase 2: [l*16+c][K] bf16 (aliases xs)

  const int n    = blockIdx.z;
  const int c0   = blockIdx.y * 16;
  const int l0   = blockIdx.x * LT;
  const int lcnt = (NL - l0 < LT) ? (NL - l0) : LT;
  const int scnt = lcnt * NV;
  const int nct  = (scnt + 15) >> 4;

  const int tid  = threadIdx.x;
  const int lane = tid & 31;
  const int wave = tid >> 5;
  const int lhal = lane >> 4;
  const int llow = lane & 15;

  const unsigned sb = to_lds(smem);

  // ---- wave 0 issues 4 TDM descriptors: x panel (scnt x 128) + 3 W partition tiles ----
  if (wave == 0) {
    tdm_load_2d(sb, xT + ((size_t)n * SP + l0 * NV) * CIN, (unsigned)scnt, CIN);
#pragma unroll
    for (int p = 0; p < NP; ++p)
      tdm_load_2d(sb + (unsigned)(WSOFF * 2 + p * 16 * (WSP * 2)),
                  Wb + (p * COUT + c0) * CIN, 16u, CIN);
  }
  // ---- all waves: Ahat via async-to-LDS (ASYNCcnt path) ----
  for (int i = tid; i < 32 * 12; i += 256) {
    int col = i / 12, part = i % 12;
    async_copy_b128(sb + (unsigned)(AHOFF * 2 + col * (AHP * 2) + part * 16),
                    AhatG + col * 96 + part * 8);
  }
  wait_async0();
  __builtin_amdgcn_s_wait_tensorcnt(0);
  __syncthreads();

  // ---- phase 1: a[p,c,s] = W_p @ x  (barrier-free K loop over staged panel) ----
  v8f acc[4][3];
#pragma unroll
  for (int a = 0; a < 4; ++a)
#pragma unroll
    for (int p = 0; p < NP; ++p)
#pragma unroll
      for (int r = 0; r < 8; ++r) acc[a][p][r] = 0.f;

#pragma unroll
  for (int kk = 0; kk < 4; ++kk) {
#pragma unroll
    for (int ti = 0; ti < 4; ++ti) {
      int t = wave + ti * 8;
      if (t < nct) {                             // wave-uniform guard
        const u16_t* bp = &xs[(t * 16 + llow) * XSP + kk * 32 + lhal * 8];
        v16bf bf = ld_frag2(bp, bp + 16);
#pragma unroll
        for (int p = 0; p < NP; ++p) {
          const u16_t* ap = &wst[(p * 16 + llow) * WSP + kk * 32 + lhal * 8];
          v16bf af = ld_frag2(ap, ap + 16);
          acc[ti][p] = __builtin_amdgcn_wmma_f32_16x16x32_bf16(
              false, af, false, bf, (short)0, acc[ti][p], false, false);
        }
      }
    }
  }
  __syncthreads();                               // xs region retires -> becomes a' (as2)

  // ---- build a' in A-fragment layout: [l][c][K=p*25+w | 75+p=bias | pad0] bf16 ----
  {
    u32_t* z = (u32_t*)as2;
    for (int i = tid; i < (LT * 16 * AHP) / 2; i += 256) z[i] = 0u;
  }
  __syncthreads();
#pragma unroll
  for (int ti = 0; ti < 4; ++ti) {
    int t = wave + ti * 8;
    int s = t * 16 + llow;
    if (t < nct && s < scnt) {
      int l = s / NV, w = s % NV;
      int rb = lhal * 8;
#pragma unroll
      for (int p = 0; p < NP; ++p)
#pragma unroll
        for (int r = 0; r < 8; ++r)
          as2[(l * 16 + rb + r) * AHP + p * NV + w] = f2bf(acc[ti][p][r]);
    }
  }
  for (int i = tid; i < LT * 16 * NP; i += 256) {
    int l = i / 48, c = (i % 48) / 3, p = i % 3;
    as2[(l * 16 + c) * AHP + 75 + p] = f2bf(bias[p * COUT + c0 + c]);
  }
  __syncthreads();

  // ---- phase 2: per-l 16x25 GEMM, K=96 (3 WMMA k-steps), e = a' @ Ahat ----
#pragma unroll
  for (int lj = 0; lj < 2; ++lj) {
    int l = wave + lj * 8;
    if (l < lcnt) {                              // wave-uniform guard
#pragma unroll
      for (int nt = 0; nt < 2; ++nt) {
        v8f d;
#pragma unroll
        for (int r = 0; r < 8; ++r) d[r] = 0.f;
#pragma unroll
        for (int kk = 0; kk < 3; ++kk) {
          const u16_t* ap = &as2[(l * 16 + llow) * AHP + kk * 32 + lhal * 8];
          v16bf af = ld_frag2(ap, ap + 16);
          const u16_t* bp = &ahat[(nt * 16 + llow) * AHP + kk * 32 + lhal * 8];
          v16bf bf = ld_frag2(bp, bp + 16);
          d = __builtin_amdgcn_wmma_f32_16x16x32_bf16(
              false, af, false, bf, (short)0, d, false, false);
        }
        int v = nt * 16 + llow;
        if (v < NV) {
#pragma unroll
          for (int r = 0; r < 8; ++r) {
            int c = r + lhal * 8;
            e[((size_t)n * COUT + c0 + c) * SP + (size_t)(l0 + l) * NV + v] = d[r];
          }
        }
      }
    }
  }
}

// ---------- kernel 2: causal 9-tap temporal sum + residual + ReLU (e is L2-resident) ----------

__global__ __launch_bounds__(256)
void stgcn_temporal_kernel(const float* __restrict__ e,
                           const float* __restrict__ x,
                           float* __restrict__ out, int total) {
  int idx = blockIdx.x * 256 + threadIdx.x;
  if (idx >= total) return;
  int l = (idx / NV) % NL;
  float s = 0.f;
#pragma unroll
  for (int j = 0; j < 9; ++j)
    if (l >= j) s += e[idx - j * NV];
  float r = s + x[idx];
  __builtin_nontemporal_store(r > 0.f ? r : 0.f, &out[idx]);
}

// ---------- launch ----------

extern "C" void kernel_launch(void* const* d_in, const int* in_sizes, int n_in,
                              void* d_out, int out_size, void* d_ws, size_t ws_size,
                              hipStream_t stream) {
  const float* x = (const float*)d_in[0];
  const float* A = (const float*)d_in[1];
  const float* W = (const float*)d_in[2];
  const float* b = (const float*)d_in[3];
  float* out = (float*)d_out;

  // workspace: [Wb 98304B][AhatG 6144B][xT 61,440,000B][e 122,880,000B]
  char* ws = (char*)d_ws;
  u16_t* Wb    = (u16_t*)ws;
  u16_t* AhatG = (u16_t*)(ws + 98304);
  u16_t* xT    = (u16_t*)(ws + 98304 + 6144);
  float* e     = (float*)(ws + 98304 + 6144 + (size_t)NB * SP * CIN * 2);

  const int wcount = COUT * NP * CIN;            // 49152
  convw_kernel<<<(wcount + 255) / 256, 256, 0, stream>>>(W, Wb, wcount);
  ahat_kernel<<<12, 256, 0, stream>>>(A, AhatG);
  xpose_kernel<<<dim3((SP + 63) / 64, NB), 256, 0, stream>>>(x, xT);

  dim3 g1((NL + LT - 1) / LT, COUT / 16, NB);    // 19 x 8 x 32
  stgcn_gemm_kernel<<<g1, 256, 0, stream>>>(xT, Wb, AhatG, b, e);

  const int total = NB * COUT * NL * NV;         // 30,720,000
  stgcn_temporal_kernel<<<(total + 255) / 256, 256, 0, stream>>>(e, x, out, total);
}
